// Expert_11020886081858
// MI455X (gfx1250) — matrix operand order repaired and checked by reading, not compiled, not run on hardware
//
#include <hip/hip_runtime.h>

typedef __attribute__((ext_vector_type(16))) __bf16 v16bf;
typedef __attribute__((ext_vector_type(8)))  float  v8f;

// ---------------------------------------------------------------------------
// Problem constants (from reference): B=4 S=2048 -> T=8192 tokens,
// D=4096, DO=4096, E=8, R=16, SCALING = 16/16 = 1.0
// ---------------------------------------------------------------------------
#define D_DIM   4096
#define DO_DIM  4096
#define E_NUM   8
#define R_NUM   16

// w1: bf16, layout [9 tiles][128 kb][32 lane][16 j]   (tiles 0..7 = A_e, 8 = route)
#define W1_ELEMS (9 * 128 * 32 * 16)
// w2: bf16, layout [256 ntile][4 kb2][32 lane][16 j]  (K = E*R = 128)
#define W2_ELEMS (256 * 4 * 32 * 16)

__device__ __forceinline__ unsigned short f2bf(float f) {
  unsigned u = __float_as_uint(f);
  u += 0x7FFFu + ((u >> 16) & 1u);     // round to nearest even
  return (unsigned short)(u >> 16);
}

__device__ __forceinline__ v8f wmma_bf16(v16bf a, v16bf b, v8f c) {
  return __builtin_amdgcn_wmma_f32_16x16x32_bf16(
      /*neg_a=*/false, a, /*neg_b=*/false, b,
      /*c_mod=*/(short)0, c, /*reuse_a=*/false, /*reuse_b=*/false);
}

// ---------------------------------------------------------------------------
// Pack A (f32 [E,R,D]) + route_w (f32 [E,D]) into bf16 WMMA B-fragments.
// B-fragment (32x16 bf16): lane -> N = lane&15 ; K = kb*32 + (j&7) + 8*(lane>>4) + 16*(j>>3)
// ---------------------------------------------------------------------------
__global__ void pack_w1(const float* __restrict__ A,
                        const float* __restrict__ route_w,
                        unsigned short* __restrict__ w1) {
  int idx = blockIdx.x * blockDim.x + threadIdx.x;
  if (idx >= W1_ELEMS) return;
  int j    = idx & 15;
  int lane = (idx >> 4) & 31;
  int kb   = (idx >> 9) & 127;
  int tile = idx >> 16;
  int n = lane & 15;
  int k = kb * 32 + (j & 7) + ((lane >> 4) << 3) + ((j >> 3) << 4);
  float v;
  if (tile < E_NUM)
    v = A[(size_t)(tile * R_NUM + n) * D_DIM + k];
  else
    v = (n < E_NUM) ? route_w[(size_t)n * D_DIM + k] : 0.0f;
  w1[idx] = f2bf(v);
}

// ---------------------------------------------------------------------------
// Pack Bw (f32 [E,DO,R]) into bf16 B-fragments for the K=128 up-projection.
// er = e*16 + r ; fragment k index = kb2*32 + (j&7) + 8*(lane>>4) + 16*(j>>3)
// ---------------------------------------------------------------------------
__global__ void pack_w2(const float* __restrict__ Bw,
                        unsigned short* __restrict__ w2) {
  int idx = blockIdx.x * blockDim.x + threadIdx.x;
  if (idx >= W2_ELEMS) return;
  int j    = idx & 15;
  int lane = (idx >> 4) & 31;
  int kb2  = (idx >> 9) & 3;
  int nt   = idx >> 11;
  int o  = nt * 16 + (lane & 15);
  int er = kb2 * 32 + (j & 7) + ((lane >> 4) << 3) + ((j >> 3) << 4);
  int e = er >> 4, r = er & 15;
  w2[idx] = f2bf(Bw[((size_t)e * DO_DIM + o) * R_NUM + r]);
}

// ---------------------------------------------------------------------------
// Main fused kernel: one block per 16-token tile, 8 waves (wave32).
// Branchless GEMM1 (every wave also accumulates router logits), in-wave
// shuffle argmax, software-pipelined WMMA loops, next-chunk prefetch.
// ---------------------------------------------------------------------------
__global__ __launch_bounds__(256) void moe_lora_kernel(
    const float* __restrict__ x,
    const unsigned short* __restrict__ w1,
    const unsigned short* __restrict__ w2,
    float* __restrict__ out) {
  __shared__ unsigned short xlds[32 * 512];   // 32 KB: x chunk, A-fragment layout
  __shared__ unsigned short whlds[4 * 512];   // 4 KB : wh A-fragments (K=128)

  const int tid  = threadIdx.x;
  const int wv   = tid >> 5;       // wave id 0..7 (== expert id)
  const int lane = tid & 31;
  const int t0   = blockIdx.x * 16;
  const float* xbase = x + (size_t)t0 * D_DIM;

  v8f acc  = {};   // h_e tile (16 tokens x 16 r), f32 accum
  v8f acc2 = {};   // router logits tile (computed redundantly by all waves)

  // ---- Phase 1+2: stream x in 4 chunks of K=1024, stage->LDS, accumulate ---
  for (int c = 0; c < 4; ++c) {
    // stage chunk c: 16 rows x 1024 cols f32 -> bf16 permuted A-fragment layout
    for (int i = 0; i < 16; ++i) {
      int idx4 = i * 256 + tid;            // float4 index within chunk
      int m  = idx4 >> 8;                  // 256 float4 per row
      int kk = (idx4 & 255) << 2;          // k offset within chunk (mult of 4)
      float4 v = *(const float4*)(xbase + (size_t)m * D_DIM + c * 1024 + kk);
      int kb = kk >> 5, ko = kk & 31;
      // 16-bit A-fragment permutation: lane = m + 16*bit3(ko), j = (ko&7)+8*bit4(ko)
      int lp = m + (((ko >> 3) & 1) << 4);
      int jp = (ko & 7) + (((ko >> 4) & 1) << 3);
      unsigned a0 = (unsigned)f2bf(v.x) | ((unsigned)f2bf(v.y) << 16);
      unsigned a1 = (unsigned)f2bf(v.z) | ((unsigned)f2bf(v.w) << 16);
      unsigned long long pk = (unsigned long long)a0 | ((unsigned long long)a1 << 32);
      *(unsigned long long*)(xlds + kb * 512 + lp * 16 + jp) = pk;   // ds_store_b64
    }
    __syncthreads();

    // prefetch next x chunk into cache while this chunk's WMMAs run
    if (c < 3) {
      for (int i = 0; i < 2; ++i) {
        int l = i * 256 + tid;             // 512 x 128B lines per chunk
        int m = l >> 5, off = (l & 31) << 5;
        __builtin_prefetch(xbase + (size_t)m * D_DIM + (c + 1) * 1024 + off, 0, 3);
      }
    }

    // GEMM1 K-loop, software-pipelined; 2 WMMAs/step, no divergence
    const unsigned short* wB = w1 + ((size_t)(wv * 128 + c * 32)) * 512 + lane * 16;
    const unsigned short* wR = w1 + ((size_t)(8 * 128 + c * 32)) * 512 + lane * 16;
    v16bf a  = *(const v16bf*)(xlds + lane * 16);
    v16bf b  = *(const v16bf*)(wB);
    v16bf br = *(const v16bf*)(wR);
    for (int kb = 0; kb < 31; ++kb) {
      v16bf an  = *(const v16bf*)(xlds + (kb + 1) * 512 + lane * 16);
      v16bf bn  = *(const v16bf*)(wB + (size_t)(kb + 1) * 512);
      v16bf brn = *(const v16bf*)(wR + (size_t)(kb + 1) * 512);
      acc  = wmma_bf16(a, b,  acc);
      acc2 = wmma_bf16(a, br, acc2);
      a = an; b = bn; br = brn;
    }
    acc  = wmma_bf16(a, b,  acc);
    acc2 = wmma_bf16(a, br, acc2);
    __syncthreads();
  }

  // ---- Phase 3: per-wave top-1 routing via shuffle butterflies -------------
  // acc2 C layout: vgpr v, lanes 0-15 -> (t=v, e=lane); lanes 16-31 -> (t=8+v, e=lane-16)
  float scales[8];
  {
    int e = lane & 15;
    for (int v = 0; v < 8; ++v) {
      float val = (e < E_NUM) ? acc2[v] : -3.402823466e38f;  // cols 8..15 are pad
      int idx = e;
      for (int m = 1; m < 16; m <<= 1) {      // reduce within each 16-lane half
        float ov = __shfl_xor(val, m, 32);
        int   oi = __shfl_xor(idx, m, 32);
        if (ov > val) { val = ov; idx = oi; }
      }
      // raw top logit kept as routing weight (reference); SCALING == 1.0
      scales[v] = (idx == wv) ? val : 0.0f;
    }
  }

  // ---- Phase 4: scale h by route weight, scatter to wh A-fragment (K=128) --
  {
    int r   = lane & 15;
    int lp  = (r >> 3) & 1;                 // er = wv*16 + r ; ko = (wv&1)*16 + r
    int jp  = (r & 7) + ((wv & 1) << 3);
    int kb2 = wv >> 1;
    int tb  = (lane >> 4) << 3;
    for (int v = 0; v < 8; ++v) {
      int t = v + tb;
      whlds[kb2 * 512 + (t + (lp << 4)) * 16 + jp] = f2bf(acc[v] * scales[v]);
    }
  }
  __syncthreads();

  // ---- Phase 5: out[16 x 4096] = wh[16 x 128] * Bw2[128 x 4096] ------------
  v16bf aw0 = *(const v16bf*)(whlds + 0 * 512 + lane * 16);
  v16bf aw1 = *(const v16bf*)(whlds + 1 * 512 + lane * 16);
  v16bf aw2 = *(const v16bf*)(whlds + 2 * 512 + lane * 16);
  v16bf aw3 = *(const v16bf*)(whlds + 3 * 512 + lane * 16);
  float* obase = out + (size_t)t0 * DO_DIM;

  const unsigned short* w2w = w2 + (size_t)(wv * 32) * 2048 + lane * 16;
  v16bf b0 = *(const v16bf*)(w2w + 0 * 512);
  v16bf b1 = *(const v16bf*)(w2w + 1 * 512);
  v16bf b2 = *(const v16bf*)(w2w + 2 * 512);
  v16bf b3 = *(const v16bf*)(w2w + 3 * 512);
  for (int i = 0; i < 32; ++i) {
    v16bf n0 = b0, n1 = b1, n2 = b2, n3 = b3;
    if (i < 31) {
      const unsigned short* np = w2w + (size_t)(i + 1) * 2048;
      n0 = *(const v16bf*)(np + 0 * 512);
      n1 = *(const v16bf*)(np + 1 * 512);
      n2 = *(const v16bf*)(np + 2 * 512);
      n3 = *(const v16bf*)(np + 3 * 512);
    }
    v8f cacc = {};
    cacc = wmma_bf16(aw0, b0, cacc);
    cacc = wmma_bf16(aw1, b1, cacc);
    cacc = wmma_bf16(aw2, b2, cacc);
    cacc = wmma_bf16(aw3, b3, cacc);
    int tile = wv * 32 + i;
    int col = tile * 16 + (lane & 15);
    int tb  = (lane >> 4) << 3;
    for (int v = 0; v < 8; ++v)
      obase[(size_t)(tb + v) * DO_DIM + col] = cacc[v];
    b0 = n0; b1 = n1; b2 = n2; b3 = n3;
  }
}

// ---------------------------------------------------------------------------
extern "C" void kernel_launch(void* const* d_in, const int* in_sizes, int n_in,
                              void* d_out, int out_size, void* d_ws, size_t ws_size,
                              hipStream_t stream) {
  const float* x       = (const float*)d_in[0];   // [4,2048,4096]
  const float* route_w = (const float*)d_in[1];   // [8,4096]
  const float* A       = (const float*)d_in[2];   // [8,16,4096]
  const float* Bw      = (const float*)d_in[3];   // [8,4096,16]
  float* out = (float*)d_out;

  unsigned short* w1 = (unsigned short*)d_ws;
  unsigned short* w2 = (unsigned short*)((char*)d_ws + (size_t)W1_ELEMS * 2);

  pack_w1<<<(W1_ELEMS + 255) / 256, 256, 0, stream>>>(A, route_w, w1);
  pack_w2<<<(W2_ELEMS + 255) / 256, 256, 0, stream>>>(Bw, w2);

  // 8192 tokens / 16 per block = 512 blocks
  moe_lora_kernel<<<512, 256, 0, stream>>>(x, w1, w2, out);
}
